// GATLayer_11828339933793
// MI455X (gfx1250) — matrix operand order repaired and checked by reading, not compiled, not run on hardware
//
#include <hip/hip_runtime.h>
#include <hip/hip_bf16.h>
#include <math.h>

// ---------------------------------------------------------------------------
// GAT layer for MI455X (gfx1250, wave32, WMMA).
//   N = 100000 nodes, D_IN = 256, H = 4 heads, F = 32 (H*F = 128), E = 1.6M.
// ---------------------------------------------------------------------------

typedef __attribute__((ext_vector_type(16))) __bf16 v16bf;
typedef __attribute__((ext_vector_type(8)))  float  v8f;

#define D_IN   256
#define HF     128   // N_HEADS * D_OUT
#define NHEAD  4
#define FDIM   32
#define NEG_SLOPE 0.2f
#define TILE_M 64    // rows per block in the GEMM

// ---------------------------------------------------------------------------
// Kernel 1: ft = x @ fc_w^T  via v_wmma_f32_16x16x32_bf16 (fp32 accumulate).
//   Block = 256 threads = 8 waves. Wave w owns output N-tile (cols w*16..+15).
//   A tile (64 rows x 256 K) staged in LDS as bf16; B (fc_w) streamed through
//   cache (128 KB, L2-hot) and converted to bf16 in registers.
//   Bounds checks are block-uniform: only the single tail block (100000 =
//   64*1562 + 32) takes the guarded path, so the hot path is branch-free.
// ---------------------------------------------------------------------------
__global__ __launch_bounds__(256) void gat_gemm_ft(const float* __restrict__ x,
                                                   const float* __restrict__ fc_w,
                                                   float* __restrict__ ft,
                                                   int n_nodes)
{
    __shared__ __bf16 lA[TILE_M * D_IN];   // 32 KB

    const int t    = threadIdx.x;
    const int wave = t >> 5;               // 0..7 -> N tile
    const int lane = t & 31;
    const int g    = lane >> 4;            // half-wave group
    const int ln   = lane & 15;
    const int m0   = blockIdx.x * TILE_M;
    const bool full = (m0 + TILE_M) <= n_nodes;   // block-uniform predicate

    // Stage A tile into LDS (bf16). Consecutive threads -> consecutive K.
    if (full) {
        #pragma unroll
        for (int i = t; i < TILE_M * D_IN; i += 256) {
            const int r = i >> 8;
            const int k = i & (D_IN - 1);
            lA[i] = (__bf16)x[(size_t)(m0 + r) * D_IN + k];
        }
    } else {
        for (int i = t; i < TILE_M * D_IN; i += 256) {
            const int r = i >> 8;
            const int k = i & (D_IN - 1);
            const int gr = m0 + r;
            const float v = (gr < n_nodes) ? x[(size_t)gr * D_IN + k] : 0.0f;
            lA[i] = (__bf16)v;
        }
    }
    __syncthreads();

    const int bcol = wave * 16 + ln;       // output column j = fc_w row index

    v8f acc[4];
    #pragma unroll
    for (int mt = 0; mt < 4; ++mt) acc[mt] = (v8f){0.f,0.f,0.f,0.f,0.f,0.f,0.f,0.f};

    for (int kc = 0; kc < D_IN; kc += 32) {
        // B fragment: B[k][n] = fc_w[n][k]; lane holds N=ln, element e -> K = e + 16*g.
        v16bf bfrag;
        const float* bptr = fc_w + (size_t)bcol * D_IN + kc + 16 * g;
        #pragma unroll
        for (int e = 0; e < 16; ++e) bfrag[e] = (__bf16)bptr[e];

        #pragma unroll
        for (int mt = 0; mt < 4; ++mt) {
            // A fragment: lane holds row M = ln; element e -> K = e + 16*g (matches B).
            const __bf16* aptr = &lA[(mt * 16 + ln) * D_IN + kc];
            v16bf afrag;
            #pragma unroll
            for (int e = 0; e < 16; ++e) afrag[e] = aptr[e + 16 * g];

            acc[mt] = __builtin_amdgcn_wmma_f32_16x16x32_bf16(
                false, afrag, false, bfrag, (short)0, acc[mt], false, false);
        }
    }

    // D layout: lane holds column N=ln; VGPR r holds row M = r + 8*g.
    float* fbase = ft + (size_t)m0 * HF + wave * 16 + ln;
    if (full) {
        #pragma unroll
        for (int mt = 0; mt < 4; ++mt) {
            #pragma unroll
            for (int r = 0; r < 8; ++r) {
                fbase[(size_t)(mt * 16 + r + 8 * g) * HF] = acc[mt][r];
            }
        }
    } else {
        for (int mt = 0; mt < 4; ++mt) {
            for (int r = 0; r < 8; ++r) {
                const int grow = m0 + mt * 16 + r + 8 * g;
                if (grow < n_nodes)
                    ft[(size_t)grow * HF + wave * 16 + ln] = acc[mt][r];
            }
        }
    }
}

// ---------------------------------------------------------------------------
// Kernel 2: el[n,h] = <ft[n,h,:], attn_l[h,:]>, same for er. Thread per (n,h).
// ---------------------------------------------------------------------------
__global__ __launch_bounds__(256) void gat_node_logits(const float* __restrict__ ft,
                                                       const float* __restrict__ attn_l,
                                                       const float* __restrict__ attn_r,
                                                       float* __restrict__ el,
                                                       float* __restrict__ er,
                                                       int n_nodes)
{
    const int idx = blockIdx.x * blockDim.x + threadIdx.x;
    if (idx >= n_nodes * NHEAD) return;
    const int node = idx >> 2;
    const int h    = idx & 3;
    const float* f  = ft + (size_t)node * HF + h * FDIM;
    const float* al = attn_l + h * FDIM;
    const float* ar = attn_r + h * FDIM;
    float sl = 0.f, sr = 0.f;
    #pragma unroll
    for (int i = 0; i < FDIM; ++i) { const float v = f[i]; sl += v * al[i]; sr += v * ar[i]; }
    el[idx] = sl;
    er[idx] = sr;
}

// ---------------------------------------------------------------------------
// Kernel 3: init out=0, m=-inf, denom=0 (harness poisons buffers with 0xAA).
// ---------------------------------------------------------------------------
__global__ __launch_bounds__(256) void gat_init(float* __restrict__ out,
                                                float* __restrict__ m,
                                                float* __restrict__ denom,
                                                int n_nodes)
{
    const int i = blockIdx.x * blockDim.x + threadIdx.x;
    if (i < n_nodes * HF) out[i] = 0.0f;
    if (i < n_nodes * NHEAD) { m[i] = -INFINITY; denom[i] = 0.0f; }
}

// Float atomic-max via signed/unsigned bit ordering (memory init'd to -inf).
__device__ __forceinline__ void atomicMaxF(float* addr, float val)
{
    if (val >= 0.0f) atomicMax((int*)addr, __float_as_int(val));
    else             atomicMin((unsigned int*)addr, __float_as_uint(val));
}

// ---------------------------------------------------------------------------
// Kernel 4: edge scores e = leaky_relu(el[src]+er[dst]); segment max into m.
// Thread per edge, 4 heads vectorized (el/er rows are float4-aligned).
// ---------------------------------------------------------------------------
__global__ __launch_bounds__(256) void gat_edge_score(const int* __restrict__ src,
                                                      const int* __restrict__ dst,
                                                      const float* __restrict__ el,
                                                      const float* __restrict__ er,
                                                      float* __restrict__ eval,
                                                      float* __restrict__ m,
                                                      int n_edges)
{
    const int i = blockIdx.x * blockDim.x + threadIdx.x;
    if (i >= n_edges) return;
    const int s = src[i];
    const int d = dst[i];
    const float4 l = *(const float4*)(el + (size_t)s * NHEAD);
    const float4 r = *(const float4*)(er + (size_t)d * NHEAD);
    float v[4] = { l.x + r.x, l.y + r.y, l.z + r.z, l.w + r.w };
    float* ev = eval + (size_t)i * NHEAD;
    float* mm = m + (size_t)d * NHEAD;
    #pragma unroll
    for (int h = 0; h < NHEAD; ++h) {
        float e = v[h];
        e = (e > 0.0f) ? e : NEG_SLOPE * e;
        ev[h] = e;
        atomicMaxF(mm + h, e);
    }
}

// ---------------------------------------------------------------------------
// Kernel 5: ex = exp(e - m[dst]); denom[dst] += ex. Thread per edge.
// ---------------------------------------------------------------------------
__global__ __launch_bounds__(256) void gat_edge_exp(const int* __restrict__ dst,
                                                    const float* __restrict__ m,
                                                    float* __restrict__ eval,
                                                    float* __restrict__ denom,
                                                    int n_edges)
{
    const int i = blockIdx.x * blockDim.x + threadIdx.x;
    if (i >= n_edges) return;
    const int d = dst[i];
    const float4 mm = *(const float4*)(m + (size_t)d * NHEAD);
    float* ev = eval + (size_t)i * NHEAD;
    float* dn = denom + (size_t)d * NHEAD;
    const float e0 = __expf(ev[0] - mm.x);
    const float e1 = __expf(ev[1] - mm.y);
    const float e2 = __expf(ev[2] - mm.z);
    const float e3 = __expf(ev[3] - mm.w);
    ev[0] = e0; ev[1] = e1; ev[2] = e2; ev[3] = e3;
    atomicAdd(dn + 0, e0);
    atomicAdd(dn + 1, e1);
    atomicAdd(dn + 2, e2);
    atomicAdd(dn + 3, e3);
}

// ---------------------------------------------------------------------------
// Kernel 6: out[dst] += (ex/denom[dst]) * ft[src]. One wave per edge; each of
// 32 lanes owns 4 contiguous floats of the 128-wide row (coalesced 512B reads
// of ft[src], f32 atomics into out — both L2-resident: ft+out = 102 MB < 192 MB).
// ---------------------------------------------------------------------------
__global__ __launch_bounds__(256) void gat_aggregate(const int* __restrict__ src,
                                                     const int* __restrict__ dst,
                                                     const float* __restrict__ ft,
                                                     const float* __restrict__ eval,
                                                     const float* __restrict__ denom,
                                                     float* __restrict__ out,
                                                     int n_edges)
{
    const int gid  = blockIdx.x * blockDim.x + threadIdx.x;
    const int edge = gid >> 5;
    const int lane = gid & 31;
    if (edge >= n_edges) return;
    const int s = src[edge];
    const int d = dst[edge];
    const int h = lane >> 3;                 // 4 floats per lane -> same head
    const float a = eval[(size_t)edge * NHEAD + h] / denom[(size_t)d * NHEAD + h];
    const int base = lane * 4;
    const float4 f = *(const float4*)(ft + (size_t)s * HF + base);
    float* o = out + (size_t)d * HF + base;
    atomicAdd(o + 0, a * f.x);
    atomicAdd(o + 1, a * f.y);
    atomicAdd(o + 2, a * f.z);
    atomicAdd(o + 3, a * f.w);
}

// ---------------------------------------------------------------------------
extern "C" void kernel_launch(void* const* d_in, const int* in_sizes, int n_in,
                              void* d_out, int out_size, void* d_ws, size_t ws_size,
                              hipStream_t stream)
{
    const float* x      = (const float*)d_in[0];   // [N, 256]
    const float* fc_w   = (const float*)d_in[1];   // [128, 256]
    const float* attn_l = (const float*)d_in[2];   // [4, 32]
    const float* attn_r = (const float*)d_in[3];   // [4, 32]
    const int*   src    = (const int*)d_in[4];     // [E]
    const int*   dst    = (const int*)d_in[5];     // [E]
    float*       out    = (float*)d_out;           // [N, 4, 32]

    const int n_nodes = in_sizes[0] / D_IN;
    const int n_edges = in_sizes[4];

    // Workspace layout (floats), all offsets 16B aligned.
    float* ws    = (float*)d_ws;
    float* ft    = ws;                                   // N * 128
    float* el    = ft + (size_t)n_nodes * HF;            // N * 4
    float* er    = el + (size_t)n_nodes * NHEAD;         // N * 4
    float* m     = er + (size_t)n_nodes * NHEAD;         // N * 4
    float* denom = m  + (size_t)n_nodes * NHEAD;         // N * 4
    float* eval  = denom + (size_t)n_nodes * NHEAD;      // E * 4

    // 1. WMMA projection: ft = x @ fc_w^T
    gat_gemm_ft<<<(n_nodes + TILE_M - 1) / TILE_M, 256, 0, stream>>>(x, fc_w, ft, n_nodes);

    // 2. Per-node attention logits
    {
        const int total = n_nodes * NHEAD;
        gat_node_logits<<<(total + 255) / 256, 256, 0, stream>>>(ft, attn_l, attn_r, el, er, n_nodes);
    }

    // 3. Init accumulators (out=0, m=-inf, denom=0)
    {
        const int total = n_nodes * HF;
        gat_init<<<(total + 255) / 256, 256, 0, stream>>>(out, m, denom, n_nodes);
    }

    // 4. Edge scores + segment max
    gat_edge_score<<<(n_edges + 255) / 256, 256, 0, stream>>>(src, dst, el, er, eval, m, n_edges);

    // 5. exp + segment sum
    gat_edge_exp<<<(n_edges + 255) / 256, 256, 0, stream>>>(dst, m, eval, denom, n_edges);

    // 6. Weighted scatter aggregation (wave per edge)
    {
        const long long total = (long long)n_edges * 32;
        gat_aggregate<<<(unsigned)((total + 255) / 256), 256, 0, stream>>>(
            src, dst, ft, eval, denom, out, n_edges);
    }
}
